// WindowAttention_807453852274
// MI455X (gfx1250) — matrix operand order repaired and checked by reading, not compiled, not run on hardware
//
#include <hip/hip_runtime.h>
#include <hip/hip_bf16.h>

// Fused Swin window attention for gfx1250 (MI455X).
// One workgroup per window (49 tokens, dim 256, 8 heads x 32).
// - All matmuls: v_wmma_f32_16x16x32_bf16, fp32 accumulate.
// - x tile staged via global_load_async_to_lds_b128 (ASYNCcnt).
// - v stored transposed in LDS -> packed b32 stores + contiguous B-frag loads.
// - Operand tiles padded to 64 rows -> no guard branches in inner loops.
// - Softmax fully in registers via __shfl_xor (wave32).

#define NTOK 49
#define NPAD 64
#define CDIM 256
#define NH   8
#define HD   32
#define QKVD 768
#define QKW  512              // q|k row width in LDS (token-major)
#define NWIN_MASK 64

typedef __bf16 bf16_t;
typedef __attribute__((ext_vector_type(16))) __bf16 v16bf;
typedef __attribute__((ext_vector_type(8)))  float  v8f;

static __device__ __forceinline__ v8f wmma_bf16(v16bf a, v16bf b, v8f c) {
  // D = A(16x32) * B(32x16) + C(16x16), fp32 accumulate.
  return __builtin_amdgcn_wmma_f32_16x16x32_bf16(
      /*neg_a=*/false, a, /*neg_b=*/false, b,
      /*c_mod=*/(short)0, c, /*reuse_a=*/false, /*reuse_b=*/false);
}

union bfpack { bf16_t h[2]; unsigned int u; };

// ---- weight fp32 -> bf16 prepass (blocks then read bf16 from L2) ------------
__global__ void cvt_weights_kernel(const float* __restrict__ wq,
                                   const float* __restrict__ wp,
                                   bf16_t* __restrict__ dst) {
  int i = blockIdx.x * blockDim.x + threadIdx.x;
  if (i < QKVD * CDIM) dst[i] = (bf16_t)wq[i];
  if (i < CDIM * CDIM) dst[QKVD * CDIM + i] = (bf16_t)wp[i];
}

__global__ __launch_bounds__(256) void win_attn_fused(
    const float* __restrict__ x,            // [wB,49,256]
    const bf16_t* __restrict__ wqkv,        // [768,256] bf16
    const bf16_t* __restrict__ wproj,       // [256,256] bf16
    const float* __restrict__ bqkv,         // [768]
    const float* __restrict__ bproj,        // [256]
    const float* __restrict__ relb,         // [169,8]
    const unsigned char* __restrict__ mask, // [64,49,49] bool
    float* __restrict__ out)                // [wB,49,256]
{
  __shared__ __align__(16) bf16_t s_qk [NPAD * QKW];       // 65536 B q|k token-major
  __shared__ __align__(16) bf16_t s_vT [NH * HD * NPAD];   // 32768 B v transposed
  __shared__ __align__(16) bf16_t s_xao[NPAD * CDIM];      // 32768 B x-stage / attn-out
  __shared__ __align__(16) bf16_t s_at [NH * 16 * 64];     // 16384 B per-head attn strip
  __shared__ __align__(16) float  s_relb[169 * NH];        //  5408 B
  __shared__               unsigned char s_mask[NTOK * NTOK]; // 2401 B

  const int b   = blockIdx.x;
  const int tid = threadIdx.x;
  const int w   = tid >> 5;       // wave id (8 waves)
  const int l   = tid & 31;       // lane id (wave32)
  const int cb  = l & 15;         // N/col within fragment
  const int ro  = (l >> 4) * 8;   // C-frag row offset of this half-wave
  const int kh  = (l >> 4) * 8;   // unused; A-frags also take K offset k16
  const int k16 = (l >> 4) * 16;  // B-frag K offset of this half-wave

  // Prefetch next window's x into L2 (speculative, no counters).
  if (b + 1 < (int)gridDim.x) {
    const float* xn = x + (size_t)(b + 1) * (NTOK * CDIM);
    __builtin_prefetch(xn + tid * 49, 0, 0);   // 256 thr * ~196B covers 50 KB
  }

  // ---------------- Phase 1a: async-copy raw fp32 x tile into LDS -------------
  // Each lane moves 16 B per op; wave-level ASYNCcnt tracks completion.
  {
    const char* xgb = (const char*)(x + (size_t)b * (NTOK * CDIM));
    const unsigned int lbase = (unsigned int)(uintptr_t)&s_qk[0]; // LDS byte offset
    for (int off = tid * 16; off < NTOK * CDIM * 4; off += 4096) {
      unsigned long long ga = (unsigned long long)(xgb + off);
      unsigned int       la = lbase + (unsigned int)off;
      asm volatile("global_load_async_to_lds_b128 %0, %1, off"
                   :: "v"(la), "v"(ga) : "memory");
    }
    asm volatile("s_wait_asynccnt 0x0" ::: "memory");
  }
  __syncthreads();

  // ---------------- Phase 1b: convert to bf16 (padded to 64 rows); bias/mask --
  {
    const float2* xf2 = (const float2*)s_qk;          // staged fp32 x
    unsigned int* dst = (unsigned int*)s_xao;
    for (int i = tid; i < NTOK * CDIM / 2; i += 256) {
      float2 v = xf2[i];
      bfpack pk; pk.h[0] = (bf16_t)v.x; pk.h[1] = (bf16_t)v.y;
      dst[i] = pk.u;
    }
    unsigned int* zp = (unsigned int*)(s_xao + NTOK * CDIM);   // rows 49..63 -> 0
    for (int i = tid; i < (NPAD - NTOK) * CDIM / 2; i += 256) zp[i] = 0u;
    const unsigned char* mg = mask + (size_t)(b & (NWIN_MASK - 1)) * (NTOK * NTOK);
    for (int i = tid; i < NTOK * NTOK; i += 256) s_mask[i] = mg[i];
    for (int i = tid; i < 169 * NH; i += 256) s_relb[i] = relb[i];
  }
  __syncthreads();

  // ---------------- Phase 2: QKV = X @ Wqkv^T + b (all 64 rows, no guards) ----
  // q/k (cols 0..511) stored token-major in s_qk; v (cols 512..767) stored
  // transposed in s_vT so the C-store packs pairs and attn@v B-frags are
  // contiguous.
  for (int nt = w; nt < QKVD / 16; nt += NH) {
    const int n0   = nt * 16;
    const int colg = n0 + cb;
    const float bv = bqkv[colg];
    v8f acc[4];
    #pragma unroll
    for (int mt = 0; mt < 4; ++mt)
      #pragma unroll
      for (int r = 0; r < 8; ++r) acc[mt][r] = bv;

    for (int k = 0; k < CDIM / 32; ++k) {
      v16bf bf;  // B frag: lane = N, half-wave selects K 0-15 / 16-31
      {
        const bf16_t* p = wqkv + (size_t)colg * CDIM + k * 32 + k16;
        #pragma unroll
        for (int i = 0; i < 16; ++i) bf[i] = p[i];
      }
      #pragma unroll
      for (int mt = 0; mt < 4; ++mt) {
        v16bf af;  // A frag: lane = M, half-wave selects K 0-15 / 16-31
        const bf16_t* p = s_xao + (mt * 16 + cb) * CDIM + k * 32 + k16;
        #pragma unroll
        for (int i = 0; i < 16; ++i) af[i] = p[i];
        acc[mt] = wmma_bf16(af, bf, acc[mt]);
      }
    }
    if (nt < 32) {              // q|k -> token-major rows
      #pragma unroll
      for (int mt = 0; mt < 4; ++mt)
        #pragma unroll
        for (int r = 0; r < 8; ++r)
          s_qk[(mt * 16 + ro + r) * QKW + colg] = (bf16_t)acc[mt][r];
    } else {                    // v -> transposed, packed pair stores
      const int dp = colg - 2 * CDIM;            // 0..255
      bf16_t* vd = s_vT + (dp >> 5) * (HD * NPAD) + (dp & 31) * NPAD;
      #pragma unroll
      for (int mt = 0; mt < 4; ++mt)
        #pragma unroll
        for (int r = 0; r < 8; r += 2) {
          bfpack pk;
          pk.h[0] = (bf16_t)acc[mt][r];
          pk.h[1] = (bf16_t)acc[mt][r + 1];
          *(unsigned int*)(vd + mt * 16 + ro + r) = pk.u;
        }
    }
  }
  __syncthreads();

  // ---------------- Phase 3: attention; wave w == head w ----------------------
  {
    const int h = w;
    const bf16_t* qb = s_qk + h * HD;
    const bf16_t* kb = s_qk + CDIM + h * HD;
    const bf16_t* vT = s_vT + h * (HD * NPAD);
    bf16_t* at = s_at + h * (16 * 64);
    const float scale = 0.17677669529663687f;  // 1/sqrt(32)

    int jhv[4], jwv[4];
    #pragma unroll
    for (int nt = 0; nt < 4; ++nt) {
      const int colg = nt * 16 + cb;
      jhv[nt] = colg / 7; jwv[nt] = colg - jhv[nt] * 7;
    }

    for (int mt = 0; mt < 4; ++mt) {
      const int m0 = mt * 16;
      // q fragment: 16x32 (K = head_dim), single WMMA K-step
      v16bf qf;
      {
        const bf16_t* p = qb + (m0 + cb) * QKW + k16;
        #pragma unroll
        for (int i = 0; i < 16; ++i) qf[i] = p[i];
      }
      // scores: 16x64 strip in registers
      v8f acc[4];
      #pragma unroll
      for (int nt = 0; nt < 4; ++nt) {
        v16bf kf;  // B frag = rows of k (contiguous in head slice)
        const bf16_t* p = kb + (nt * 16 + cb) * QKW + k16;
        #pragma unroll
        for (int i = 0; i < 16; ++i) kf[i] = p[i];
        v8f z;
        #pragma unroll
        for (int r = 0; r < 8; ++r) z[r] = 0.f;
        acc[nt] = wmma_bf16(qf, kf, z);
      }

      // scale + relative-position bias + mask (element-wise, LDS gathers)
      #pragma unroll
      for (int nt = 0; nt < 4; ++nt) {
        const int colg = nt * 16 + cb;
        const bool colv = colg < NTOK;
        #pragma unroll
        for (int r = 0; r < 8; ++r) {
          const int i = m0 + ro + r;
          float sv = -1e30f;
          if (colv && i < NTOK) {
            const int ih = i / 7, iw = i - ih * 7;
            const int ridx = (ih - jhv[nt] + 6) * 13 + (iw - jwv[nt] + 6);
            sv = acc[nt][r] * scale + s_relb[ridx * NH + h];
            if (!s_mask[i * NTOK + colg]) sv = -100.0f;
          }
          acc[nt][r] = sv;
        }
      }

      // in-register softmax: 4 per-lane values + 16-lane half reduction
      float mx[8], sm[8];
      #pragma unroll
      for (int r = 0; r < 8; ++r) {
        float m = fmaxf(fmaxf(acc[0][r], acc[1][r]), fmaxf(acc[2][r], acc[3][r]));
        m = fmaxf(m, __shfl_xor(m, 1));
        m = fmaxf(m, __shfl_xor(m, 2));
        m = fmaxf(m, __shfl_xor(m, 4));
        m = fmaxf(m, __shfl_xor(m, 8));
        mx[r] = m;
      }
      #pragma unroll
      for (int nt = 0; nt < 4; ++nt)
        #pragma unroll
        for (int r = 0; r < 8; ++r) acc[nt][r] = __expf(acc[nt][r] - mx[r]);
      #pragma unroll
      for (int r = 0; r < 8; ++r) {
        float s = (acc[0][r] + acc[1][r]) + (acc[2][r] + acc[3][r]);
        s += __shfl_xor(s, 1);
        s += __shfl_xor(s, 2);
        s += __shfl_xor(s, 4);
        s += __shfl_xor(s, 8);
        sm[r] = 1.0f / s;
      }
      // write bf16 attn strip (cols >= 49 forced to 0 for K-padding)
      #pragma unroll
      for (int nt = 0; nt < 4; ++nt) {
        const int colg = nt * 16 + cb;
        const float cz = (colg < NTOK) ? 1.0f : 0.0f;
        #pragma unroll
        for (int r = 0; r < 8; ++r)
          at[(ro + r) * 64 + colg] = (bf16_t)(acc[nt][r] * sm[r] * cz);
      }
      // wave-private LDS write->read; dscnt ordering handled by compiler.

      // out strip = attn(16x64) @ v(64x32): 2 K-steps x 2 N-tiles
      v8f oacc[2];
      #pragma unroll
      for (int nd = 0; nd < 2; ++nd)
        #pragma unroll
        for (int r = 0; r < 8; ++r) oacc[nd][r] = 0.f;

      #pragma unroll
      for (int kt = 0; kt < 2; ++kt) {
        v16bf af;  // A frag from attn strip (row-major, contiguous)
        {
          const bf16_t* p = at + cb * 64 + kt * 32 + k16;
          #pragma unroll
          for (int i = 0; i < 16; ++i) af[i] = p[i];
        }
        #pragma unroll
        for (int nd = 0; nd < 2; ++nd) {
          v16bf vf;  // B frag from transposed v: fully contiguous 32 B
          const bf16_t* p = vT + (nd * 16 + cb) * NPAD + kt * 32 + k16;
          #pragma unroll
          for (int i = 0; i < 16; ++i) vf[i] = p[i];
          oacc[nd] = wmma_bf16(af, vf, oacc[nd]);
        }
      }
      #pragma unroll
      for (int nd = 0; nd < 2; ++nd)
        #pragma unroll
        for (int r = 0; r < 8; ++r)
          s_xao[(m0 + ro + r) * CDIM + h * HD + nd * 16 + cb] = (bf16_t)oacc[nd][r];
    }
  }
  __syncthreads();

  // ---------------- Phase 4: out = AO @ Wproj^T + b -> global fp32 ------------
  {
    float* og = out + (size_t)b * (NTOK * CDIM);
    for (int nt = w; nt < CDIM / 16; nt += NH) {
      const int n0   = nt * 16;
      const int colg = n0 + cb;
      const float bv = bproj[colg];
      v8f acc[4];
      #pragma unroll
      for (int mt = 0; mt < 4; ++mt)
        #pragma unroll
        for (int r = 0; r < 8; ++r) acc[mt][r] = bv;

      for (int k = 0; k < CDIM / 32; ++k) {
        v16bf bf;
        {
          const bf16_t* p = wproj + (size_t)colg * CDIM + k * 32 + k16;
          #pragma unroll
          for (int i = 0; i < 16; ++i) bf[i] = p[i];
        }
        #pragma unroll
        for (int mt = 0; mt < 4; ++mt) {
          v16bf af;
          const bf16_t* p = s_xao + (mt * 16 + cb) * CDIM + k * 32 + k16;
          #pragma unroll
          for (int i = 0; i < 16; ++i) af[i] = p[i];
          acc[mt] = wmma_bf16(af, bf, acc[mt]);
        }
      }
      #pragma unroll
      for (int mt = 0; mt < 4; ++mt)
        #pragma unroll
        for (int r = 0; r < 8; ++r) {
          const int i = mt * 16 + ro + r;
          if (i < NTOK) og[(size_t)i * CDIM + colg] = acc[mt][r];
        }
    }
  }
}

extern "C" void kernel_launch(void* const* d_in, const int* in_sizes, int n_in,
                              void* d_out, int out_size, void* d_ws, size_t ws_size,
                              hipStream_t stream) {
  (void)n_in; (void)out_size; (void)ws_size;
  const float* x      = (const float*)d_in[0];
  const float* w_qkv  = (const float*)d_in[1];   // [768,256]
  const float* b_qkv  = (const float*)d_in[2];   // [768]
  const float* w_proj = (const float*)d_in[3];   // [256,256]
  const float* b_proj = (const float*)d_in[4];   // [256]
  const float* relb   = (const float*)d_in[5];   // [169,8]
  const unsigned char* mask = (const unsigned char*)d_in[6]; // [64,49,49] bool
  float* out = (float*)d_out;

  const int wB = in_sizes[0] / (NTOK * CDIM);    // 4096

  bf16_t* wbf = (bf16_t*)d_ws;                   // [768*256] qkv + [256*256] proj
  cvt_weights_kernel<<<(QKVD * CDIM + 255) / 256, 256, 0, stream>>>(w_qkv, w_proj, wbf);
  win_attn_fused<<<wB, 256, 0, stream>>>(x, wbf, wbf + QKVD * CDIM,
                                         b_qkv, b_proj, relb, mask, out);
}
